// SOG_3D_pointcharge_309237645344
// MI455X (gfx1250) — compile-verified, matrix operand order checked by reading, not run on hardware
//
#include <hip/hip_runtime.h>

typedef _Float16 half_t;
typedef __attribute__((ext_vector_type(8)))  _Float16 v8h;
typedef __attribute__((ext_vector_type(16))) _Float16 v16h;
typedef __attribute__((ext_vector_type(8)))  float    v8f;

#define NB 2
#define NP 2048
#define NK 49
#define NAC 2401           // 49*49
#define NACPAD 2416        // padded to a whole 16-row tile (151*16)
#define NK3 117649         // 49^3
#define NDPAD 64
#define NBW 12
#define TWO_PI_OVER_L 0.6283185307179586f
#define HALF_L 5.0f
#define INV_2V (1.0f / 2000.0f)   // 1/(2*L^3), L=10

static __device__ __forceinline__ v16h cat16(v8h lo, v8h hi) {
  return __builtin_shufflevector(lo, hi, 0,1,2,3,4,5,6,7,8,9,10,11,12,13,14,15);
}

// ---------------------------------------------------------------------------
// Kernel 1: mult[a,c,d] filter + diagonal sum (block-reduced atomic)
// ---------------------------------------------------------------------------
__global__ void mult_kernel(const float* __restrict__ shift,
                            const float* __restrict__ amplitud,
                            float* __restrict__ mult,
                            float* __restrict__ diagAcc) {
  __shared__ float red[256];
  int t = blockIdx.x * blockDim.x + threadIdx.x;
  float m = 0.0f;
  if (t < NK3) {
    int a = t / NAC;
    int rem = t - a * NAC;
    int c = rem / NK;
    int d = rem - c * NK;
    float kx = TWO_PI_OVER_L * (float)(a - 24);
    float ky = TWO_PI_OVER_L * (float)(c - 24);
    float kz = TWO_PI_OVER_L * (float)(d - 24);
    float sq = kx * kx + ky * ky + kz * kz;
    if (!(a == 24 && c == 24 && d == 24)) {
      for (int j = 0; j < NBW; ++j) {
        float inv_var = -expf(2.0f * shift[j]);   // -1/exp(-shift)^2
        m += amplitud[j] * expf(sq * inv_var);
      }
    }
    mult[t] = m;
  }
  red[threadIdx.x] = m;
  __syncthreads();
  for (int s = 128; s > 0; s >>= 1) {
    if ((int)threadIdx.x < s) red[threadIdx.x] += red[threadIdx.x + s];
    __syncthreads();
  }
  if (threadIdx.x == 0) atomicAdd(diagAcc, red[0]);
}

// ---------------------------------------------------------------------------
// Kernel 2: per-particle phase tables.
//   E = cos(phase) - i*sin(phase), phase = (2pi/L)(pos - L/2) * kint
// k-major f16 tables (GEMM operands, q folded into Ey) + p-major tables.
// ---------------------------------------------------------------------------
__global__ void tables_kernel(const float* __restrict__ pos,
                              const float* __restrict__ charge,
                              half_t* __restrict__ EXr, half_t* __restrict__ EXi,
                              half_t* __restrict__ EYqr, half_t* __restrict__ EYqi,
                              half_t* __restrict__ EZr, half_t* __restrict__ EZi,
                              half_t* __restrict__ EzTr, half_t* __restrict__ EzTi,
                              float* __restrict__ ExTr, float* __restrict__ ExTi,
                              float* __restrict__ EyTr, float* __restrict__ EyTi) {
  int t = blockIdx.x * blockDim.x + threadIdx.x;
  if (t >= NB * NP) return;
  int b = t >> 11;
  int p = t & (NP - 1);
  float q = charge[t];
  float x0 = TWO_PI_OVER_L * (pos[t * 3 + 0] - HALF_L);
  float x1 = TWO_PI_OVER_L * (pos[t * 3 + 1] - HALF_L);
  float x2 = TWO_PI_OVER_L * (pos[t * 3 + 2] - HALF_L);
  for (int n = 0; n < NK; ++n) {
    float kk = (float)(n - 24);
    float s, c;
    // X dimension
    sincosf(x0 * kk, &s, &c);
    float er = c, ei = -s;
    size_t km = (size_t)(b * NK + n) * NP + p;
    EXr[km] = (half_t)er;
    EXi[km] = (half_t)ei;
    ExTr[(size_t)t * NK + n] = er;
    ExTi[(size_t)t * NK + n] = ei;
    // Y dimension (fold charge in for the on-the-fly T = q*Ex*Ey)
    sincosf(x1 * kk, &s, &c);
    er = c; ei = -s;
    EYqr[km] = (half_t)(q * er);
    EYqi[km] = (half_t)(q * ei);
    EyTr[(size_t)t * NK + n] = er;
    EyTi[(size_t)t * NK + n] = ei;
    // Z dimension
    sincosf(x2 * kk, &s, &c);
    er = c; ei = -s;
    EZr[km] = (half_t)er;
    EZi[km] = (half_t)ei;
    EzTr[(size_t)t * NDPAD + n] = (half_t)er;
    EzTi[(size_t)t * NDPAD + n] = (half_t)ei;
  }
  for (int n = NK; n < NDPAD; ++n) {   // zero pad K dim for stage-2 GEMM
    EzTr[(size_t)t * NDPAD + n] = (half_t)0.0f;
    EzTi[(size_t)t * NDPAD + n] = (half_t)0.0f;
  }
}

// ---------------------------------------------------------------------------
// Kernel 3 (stage 1): structure factor GEMM.
//   F[(a,c), d] = sum_p T[(a,c),p] * Ez[p,d],  T = q*Ex*Ey built on the fly.
// One wave per 16x16 output tile; M=2401 (151 tiles), N=49 (4 tiles), K=2048.
// No operand masking: garbage in clamped A-rows / B-cols only lands in
// out-of-range D rows/cols, which are stored as zeros (mu=0), including the
// G pad rows 2401..2415 that stage 2 then reads as real zeros.
// Complex GEMM as 4 WMMA accumulator chains: Fr = P1-P2, Fi = P3+P4.
// ---------------------------------------------------------------------------
__global__ void stage1_kernel(const half_t* __restrict__ EXr, const half_t* __restrict__ EXi,
                              const half_t* __restrict__ EYqr, const half_t* __restrict__ EYqi,
                              const half_t* __restrict__ EZr, const half_t* __restrict__ EZi,
                              const float* __restrict__ mult,
                              half_t* __restrict__ Gr, half_t* __restrict__ Gi) {
  const int NT_AC = 151, NT_D = 4;
  int wave = (int)((blockIdx.x * blockDim.x + threadIdx.x) >> 5);
  if (wave >= NB * NT_AC * NT_D) return;
  int b = wave / (NT_AC * NT_D);
  int rem = wave - b * (NT_AC * NT_D);
  int acTile = rem / NT_D;
  int dTile = rem - acTile * NT_D;

  int lane = threadIdx.x & 31;
  int mr = lane & 15;
  int hi = lane >> 4;

  // A operand row: (a,c) pair (clamped for addressing); B column: d (clamped)
  int ac = acTile * 16 + mr;
  int acC = (ac < NAC) ? ac : (NAC - 1);
  int a = acC / NK;
  int c = acC - a * NK;
  int d = dTile * 16 + mr;
  int dC = (d < NK) ? d : (NK - 1);

  const half_t* pxr = EXr + (size_t)(b * NK + a) * NP;
  const half_t* pxi = EXi + (size_t)(b * NK + a) * NP;
  const half_t* pyr = EYqr + (size_t)(b * NK + c) * NP;
  const half_t* pyi = EYqi + (size_t)(b * NK + c) * NP;
  const half_t* pzr = EZr + (size_t)(b * NK + dC) * NP;
  const half_t* pzi = EZi + (size_t)(b * NK + dC) * NP;

  v8f P1 = {}, P2 = {}, P3 = {}, P4 = {};
  for (int k0 = 0; k0 < NP; k0 += 32) {
    int o0 = k0 + hi * 8;          // ISA A/B f16 layout: two 8-wide K runs
    int o1 = k0 + 16 + hi * 8;
    v8h xr0 = *(const v8h*)(pxr + o0), xr1 = *(const v8h*)(pxr + o1);
    v8h xi0 = *(const v8h*)(pxi + o0), xi1 = *(const v8h*)(pxi + o1);
    v8h yr0 = *(const v8h*)(pyr + o0), yr1 = *(const v8h*)(pyr + o1);
    v8h yi0 = *(const v8h*)(pyi + o0), yi1 = *(const v8h*)(pyi + o1);
    v8h zr0 = *(const v8h*)(pzr + o0), zr1 = *(const v8h*)(pzr + o1);
    v8h zi0 = *(const v8h*)(pzi + o0), zi1 = *(const v8h*)(pzi + o1);

    // T = q*Ex*Ey (q folded in Ey tables): Tr = XrYr - XiYi ; Ti = XrYi + XiYr
    v8h tr0 = xr0 * yr0 - xi0 * yi0;
    v8h tr1 = xr1 * yr1 - xi1 * yi1;
    v8h ti0 = xr0 * yi0 + xi0 * yr0;
    v8h ti1 = xr1 * yi1 + xi1 * yr1;

    v16h ar = cat16(tr0, tr1);
    v16h ai = cat16(ti0, ti1);
    v16h br = cat16(zr0, zr1);
    v16h bi = cat16(zi0, zi1);

    P1 = __builtin_amdgcn_wmma_f32_16x16x32_f16(false, ar, false, br, (short)0, P1, false, false);
    P2 = __builtin_amdgcn_wmma_f32_16x16x32_f16(false, ai, false, bi, (short)0, P2, false, false);
    P3 = __builtin_amdgcn_wmma_f32_16x16x32_f16(false, ar, false, bi, (short)0, P3, false, false);
    P4 = __builtin_amdgcn_wmma_f32_16x16x32_f16(false, ai, false, br, (short)0, P4, false, false);
  }

  // D tile: row m = r + hi*8 -> (a,c); col n = lane&15 -> d
  int dcol = dTile * 16 + mr;
  int dmul = (dcol < NK) ? dcol : 0;
  for (int r = 0; r < 8; ++r) {
    int m = r + hi * 8;
    int acO = acTile * 16 + m;          // < NACPAD always
    bool valid = (acO < NAC) && (dcol < NK);
    float fr = P1[r] - P2[r];
    float fi = P3[r] + P4[r];
    float mu = valid ? mult[(acO < NAC ? acO : 0) * NK + dmul] : 0.0f;
    size_t gidx = (size_t)(b * NACPAD + acO) * NDPAD + dcol;
    Gr[gidx] = (half_t)(mu * fr);
    Gi[gidx] = (half_t)(mu * fi);
  }
}

// ---------------------------------------------------------------------------
// Kernel 4 (stage 2): back-projection GEMM with fused epilogue.
//   H[p,(a,c)] = sum_d G[(a,c),d] * conj(Ez)[p,d]   (WMMA, K=64 padded)
//   H2[p,a]   += H[p,(a,c)] * conj(Ey)[p,c]         (LDS float atomics)
//   out[p]     = Re( sum_a H2[p,a]*conj(Ex)[p,a] )/(2V) - q*diag, times q
// G pad rows (2401..2415) are genuine zeros, so no masking and no epilogue
// branch is needed: pad rows contribute exactly 0 (aa may reach 49, which the
// H2[16][50] padding absorbs). Lanes stay fully converged.
// ---------------------------------------------------------------------------
__global__ void stage2_kernel(const half_t* __restrict__ Gr, const half_t* __restrict__ Gi,
                              const half_t* __restrict__ EzTr, const half_t* __restrict__ EzTi,
                              const float* __restrict__ EyTr, const float* __restrict__ EyTi,
                              const float* __restrict__ ExTr, const float* __restrict__ ExTi,
                              const float* __restrict__ charge,
                              const float* __restrict__ diagAcc,
                              float* __restrict__ out) {
  __shared__ float H2r[16][50];
  __shared__ float H2i[16][50];
  int b = (int)(blockIdx.x >> 7);
  int pTile = (int)(blockIdx.x & 127);

  for (int i = threadIdx.x; i < 16 * 50; i += blockDim.x) {
    (&H2r[0][0])[i] = 0.0f;
    (&H2i[0][0])[i] = 0.0f;
  }
  __syncthreads();

  int lane = threadIdx.x & 31;
  int waveId = (int)(threadIdx.x >> 5);   // 0..3
  int mr = lane & 15;
  int hi = lane >> 4;
  int p = pTile * 16 + mr;                // B operand column -> particle

  const half_t* pzr = EzTr + (size_t)(b * NP + p) * NDPAD;
  const half_t* pzi = EzTi + (size_t)(b * NP + p) * NDPAD;
  const float* eyr = EyTr + (size_t)(b * NP + p) * NK;
  const float* eyi = EyTi + (size_t)(b * NP + p) * NK;

  for (int acTile = waveId; acTile < 151; acTile += 4) {
    int ac = acTile * 16 + mr;            // < NACPAD, pad rows are zeros
    const half_t* pgr = Gr + (size_t)(b * NACPAD + ac) * NDPAD;
    const half_t* pgi = Gi + (size_t)(b * NACPAD + ac) * NDPAD;

    v8f P1 = {}, P2 = {}, P3 = {}, P4 = {};
    for (int k0 = 0; k0 < NDPAD; k0 += 32) {
      int o0 = k0 + hi * 8;
      int o1 = k0 + 16 + hi * 8;
      v8h gr0 = *(const v8h*)(pgr + o0), gr1 = *(const v8h*)(pgr + o1);
      v8h gi0 = *(const v8h*)(pgi + o0), gi1 = *(const v8h*)(pgi + o1);
      v8h zr0 = *(const v8h*)(pzr + o0), zr1 = *(const v8h*)(pzr + o1);
      v8h zi0 = *(const v8h*)(pzi + o0), zi1 = *(const v8h*)(pzi + o1);
      v16h agr = cat16(gr0, gr1);
      v16h agi = cat16(gi0, gi1);
      v16h bzr = cat16(zr0, zr1);
      v16h bzi = cat16(zi0, zi1);
      P1 = __builtin_amdgcn_wmma_f32_16x16x32_f16(false, agr, false, bzr, (short)0, P1, false, false);
      P2 = __builtin_amdgcn_wmma_f32_16x16x32_f16(false, agi, false, bzi, (short)0, P2, false, false);
      P3 = __builtin_amdgcn_wmma_f32_16x16x32_f16(false, agi, false, bzr, (short)0, P3, false, false);
      P4 = __builtin_amdgcn_wmma_f32_16x16x32_f16(false, agr, false, bzi, (short)0, P4, false, false);
    }
    // H = G*conj(Ez): Hr = P1+P2, Hi = P3-P4 (stored Ez = (cos,-sin))
    for (int r = 0; r < 8; ++r) {
      int m = r + hi * 8;
      int acO = acTile * 16 + m;
      int aa = acO / NK;                  // may be 49 on pad rows (zero contrib)
      int cc = acO - aa * NK;
      float hr = P1[r] + P2[r];
      float hh = P3[r] - P4[r];
      float cr = eyr[cc], ci = eyi[cc];
      // H * conj(Ey): conj(Ey) = (cr, -ci)
      atomicAdd(&H2r[mr][aa], hr * cr + hh * ci);
      atomicAdd(&H2i[mr][aa], hh * cr - hr * ci);
    }
  }
  __syncthreads();

  if (threadIdx.x < 16) {
    int pl = (int)threadIdx.x;
    int pIdx = pTile * 16 + pl;
    const float* exr = ExTr + (size_t)(b * NP + pIdx) * NK;
    const float* exi = ExTi + (size_t)(b * NP + pIdx) * NK;
    float acc = 0.0f;
    for (int a2 = 0; a2 < NK; ++a2)
      acc += H2r[pl][a2] * exr[a2] + H2i[pl][a2] * exi[a2];   // Re(H2*conj(Ex))
    float diag = diagAcc[0] * INV_2V;
    float q = charge[b * NP + pIdx];
    out[b * NP + pIdx] = (acc * INV_2V - q * diag) * q;
  }
}

// ---------------------------------------------------------------------------
// Host launcher
// ---------------------------------------------------------------------------
extern "C" void kernel_launch(void* const* d_in, const int* in_sizes, int n_in,
                              void* d_out, int out_size, void* d_ws, size_t ws_size,
                              hipStream_t stream) {
  (void)in_sizes; (void)n_in; (void)out_size; (void)ws_size;
  const float* pos      = (const float*)d_in[0];
  const float* charge   = (const float*)d_in[1];
  const float* shift    = (const float*)d_in[2];
  const float* amplitud = (const float*)d_in[3];
  float* out = (float*)d_out;

  char* ws = (char*)d_ws;
  size_t off = 0;
  auto alloc = [&](size_t bytes) -> void* {
    void* pp = (void*)(ws + off);
    off = (off + bytes + 255) & ~(size_t)255;
    return pp;
  };
  float*  diagAcc = (float*)alloc(sizeof(float));
  float*  mult    = (float*)alloc((size_t)NK3 * sizeof(float));
  const size_t kmaj = (size_t)NB * NK * NP * sizeof(half_t);
  half_t* EXr  = (half_t*)alloc(kmaj);
  half_t* EXi  = (half_t*)alloc(kmaj);
  half_t* EYqr = (half_t*)alloc(kmaj);
  half_t* EYqi = (half_t*)alloc(kmaj);
  half_t* EZr  = (half_t*)alloc(kmaj);
  half_t* EZi  = (half_t*)alloc(kmaj);
  const size_t pmajh = (size_t)NB * NP * NDPAD * sizeof(half_t);
  half_t* EzTr = (half_t*)alloc(pmajh);
  half_t* EzTi = (half_t*)alloc(pmajh);
  const size_t pmajf = (size_t)NB * NP * NK * sizeof(float);
  float* ExTr = (float*)alloc(pmajf);
  float* ExTi = (float*)alloc(pmajf);
  float* EyTr = (float*)alloc(pmajf);
  float* EyTi = (float*)alloc(pmajf);
  const size_t gsz = (size_t)NB * NACPAD * NDPAD * sizeof(half_t);
  half_t* Gr = (half_t*)alloc(gsz);
  half_t* Gi = (half_t*)alloc(gsz);

  hipMemsetAsync(diagAcc, 0, sizeof(float), stream);

  mult_kernel<<<(NK3 + 255) / 256, 256, 0, stream>>>(shift, amplitud, mult, diagAcc);

  tables_kernel<<<(NB * NP + 127) / 128, 128, 0, stream>>>(
      pos, charge, EXr, EXi, EYqr, EYqi, EZr, EZi,
      EzTr, EzTi, ExTr, ExTi, EyTr, EyTi);

  // 2 b * 151 acTiles * 4 dTiles = 1208 waves, 4 waves/block -> 302 blocks
  stage1_kernel<<<302, 128, 0, stream>>>(EXr, EXi, EYqr, EYqi, EZr, EZi, mult, Gr, Gi);

  // 2 b * 128 pTiles = 256 blocks, 4 waves each
  stage2_kernel<<<256, 128, 0, stream>>>(Gr, Gi, EzTr, EzTi, EyTr, EyTi,
                                         ExTr, ExTi, charge, diagAcc, out);
}
